// TFLongformerSelfAttention_2164663517530
// MI455X (gfx1250) — compile-verified, matrix operand order checked by reading, not run on hardware
//
#include <hip/hip_runtime.h>
#include <hip/hip_bf16.h>

// Problem constants (reference: B=1, S=4096, E=768, H=12, d=64, w=256, g=64)
#define SS 4096
#define EE 768
#define HH 12
#define DD 64
#define GG 64
#define WW 256
#define NEGV -1.0e9f

typedef __attribute__((ext_vector_type(16))) __bf16 v16bf;
typedef __attribute__((ext_vector_type(8)))  float  v8f;

#define WMMA_BF16(a, b, c) \
  __builtin_amdgcn_wmma_f32_16x16x32_bf16(false, (a), false, (b), (short)0, (c), false, false)

union Frag {
  v16bf v;
  float4 q[2];
  unsigned short u[16];
};

__device__ __forceinline__ unsigned short f2bf(float x) {
  union { float f; unsigned int u; } t; t.f = x;
  unsigned int u = t.u + 0x7fffu + ((t.u >> 16) & 1u);  // RNE
  return (unsigned short)(u >> 16);
}

// Load one 16x32 bf16 A/B fragment for lane `lane` from a k-contiguous row.
// Per CDNA5 ISA 16-bit layout: element e holds K = (e>>3)*16 + (lane>>4)*8 + (e&7),
// i.e. two contiguous 16-byte chunks at k0 + (lane>>4)*8 and k0 + 16 + (lane>>4)*8.
__device__ __forceinline__ v16bf frag_load(const unsigned short* rowp, int k0, int lane) {
  Frag f;
  int h8 = (lane >> 4) << 3;
  f.q[0] = *(const float4*)(rowp + k0 + h8);
  f.q[1] = *(const float4*)(rowp + k0 + 16 + h8);
  return f.v;
}

// ---------------- conversion kernels ----------------
__global__ void cvt_f32_bf16(const float* __restrict__ src,
                             unsigned short* __restrict__ dst, int n) {
  int i = blockIdx.x * 256 + threadIdx.x;
  if (i < n) dst[i] = f2bf(src[i]);
}

// W is [k][n] row-major (hidden @ W). Write Wt[n][k] so B fragments are k-contiguous.
__global__ void cvt_weight_t(const float* __restrict__ W,
                             unsigned short* __restrict__ Wt) {
  int i = blockIdx.x * 256 + threadIdx.x;
  if (i < EE * EE) {
    int nn = i / EE, kk = i % EE;
    Wt[nn * EE + kk] = f2bf(W[kk * EE + nn]);
  }
}

// ---------------- projection GEMM: Y[h][row][d] = bf16((X@W + b) * scale) ----------------
// grid: (rows/128, 12); block: 128 (4 waves); each wave: 32 rows x 64 cols (2x4 WMMA tiles).
// Fully-unrolled k-loop with two register buffer sets (ping-pong): next step's 6 fragment
// loads are issued before the current step's 8 WMMAs; full unroll lets SSA renaming kill
// all cross-buffer copies and resolves the pipeline tail at compile time.
__global__ void __launch_bounds__(128) gemm_proj(
    const unsigned short* __restrict__ X,    // [rows][768] bf16
    const unsigned short* __restrict__ Wt,   // [768 n][768 k] bf16 (transposed)
    const float* __restrict__ bias,          // [768]
    unsigned short* __restrict__ Y,          // [12][rows][64] bf16
    int rows, float scale) {
  int lane = threadIdx.x & 31, wave = threadIdx.x >> 5;
  int m0 = blockIdx.x * 128 + wave * 32;
  if (m0 >= rows) return;                    // qg GEMM has only 64 rows
  int n0 = blockIdx.y * 64;

  v8f z = {0.f, 0.f, 0.f, 0.f, 0.f, 0.f, 0.f, 0.f};
  v8f acc[2][4] = {{z, z, z, z}, {z, z, z, z}};

  const unsigned short* xrow[2];
  xrow[0] = X + (size_t)(m0 + (lane & 15)) * EE;
  xrow[1] = xrow[0] + 16 * (size_t)EE;
  const unsigned short* wrow[4];
#pragma unroll
  for (int nt = 0; nt < 4; ++nt)
    wrow[nt] = Wt + (size_t)(n0 + nt * 16 + (lane & 15)) * EE;

  v16bf a0[2], b0[4], a1[2], b1[4];
#pragma unroll
  for (int mt = 0; mt < 2; ++mt) a0[mt] = frag_load(xrow[mt], 0, lane);
#pragma unroll
  for (int nt = 0; nt < 4; ++nt) b0[nt] = frag_load(wrow[nt], 0, lane);

#pragma unroll
  for (int k0 = 0; k0 < EE; k0 += 64) {
    // stage k0+32 into buffer 1 (always in range: last k0 = 704 -> 736 < 768)
#pragma unroll
    for (int mt = 0; mt < 2; ++mt) a1[mt] = frag_load(xrow[mt], k0 + 32, lane);
#pragma unroll
    for (int nt = 0; nt < 4; ++nt) b1[nt] = frag_load(wrow[nt], k0 + 32, lane);
    // compute on buffer 0 (k0)
#pragma unroll
    for (int mt = 0; mt < 2; ++mt)
#pragma unroll
      for (int nt = 0; nt < 4; ++nt)
        acc[mt][nt] = WMMA_BF16(a0[mt], b0[nt], acc[mt][nt]);
    // stage k0+64 into buffer 0 (compile-time skipped on final step)
    if (k0 + 64 < EE) {
#pragma unroll
      for (int mt = 0; mt < 2; ++mt) a0[mt] = frag_load(xrow[mt], k0 + 64, lane);
#pragma unroll
      for (int nt = 0; nt < 4; ++nt) b0[nt] = frag_load(wrow[nt], k0 + 64, lane);
    }
    // compute on buffer 1 (k0+32)
#pragma unroll
    for (int mt = 0; mt < 2; ++mt)
#pragma unroll
      for (int nt = 0; nt < 4; ++nt)
        acc[mt][nt] = WMMA_BF16(a1[mt], b1[nt], acc[mt][nt]);
  }

  int rbase = (lane >> 4) * 8;
#pragma unroll
  for (int mt = 0; mt < 2; ++mt) {
#pragma unroll
    for (int nt = 0; nt < 4; ++nt) {
      int col = n0 + nt * 16 + (lane & 15);
      float bv = bias[col];
      int dd = col & 63;
#pragma unroll
      for (int r = 0; r < 8; ++r) {
        int row = m0 + mt * 16 + rbase + r;
        float v = (acc[mt][nt][r] + bv) * scale;
        Y[((size_t)blockIdx.y * rows + row) * DD + dd] = f2bf(v);
      }
    }
  }
}

// ---------------- banded + global-key attention (flash-style) ----------------
// grid: (S/64 query chunks, 12 heads); block 128 (4 waves, 16 query rows each).
__global__ void __launch_bounds__(128) attn_band(
    const unsigned short* __restrict__ Q,  // [H][S][64] bf16 (pre-scaled)
    const unsigned short* __restrict__ K,  // [H][S][64]
    const unsigned short* __restrict__ V,  // [H][S][64]
    float* __restrict__ out) {             // [S][768] f32
  __shared__ __align__(16) unsigned short Vt[64][64];  // [d][key]
  __shared__ __align__(16) unsigned short P[64][64];   // [local row][key]

  int lane = threadIdx.x & 31, wave = threadIdx.x >> 5;
  int h = blockIdx.y;
  int qbase = blockIdx.x * 64;
  const unsigned short* qh = Q + (size_t)h * SS * DD;
  const unsigned short* kh = K + (size_t)h * SS * DD;
  const unsigned short* vh = V + (size_t)h * SS * DD;

  const unsigned short* qrow = qh + (size_t)(qbase + wave * 16 + (lane & 15)) * DD;
  v16bf aq0 = frag_load(qrow, 0, lane);
  v16bf aq1 = frag_load(qrow, 32, lane);

  v8f z = {0.f, 0.f, 0.f, 0.f, 0.f, 0.f, 0.f, 0.f};
  v8f o[4] = {z, z, z, z};
  float m[8], l[8];
#pragma unroll
  for (int r = 0; r < 8; ++r) { m[r] = -1e30f; l[r] = 0.f; }
  int rbase = (lane >> 4) * 8;

  // t = -1: global key tile (keys 0..63, unmasked). t = 0..8: band tiles.
  for (int t = -1; t <= 8; ++t) {
    bool global_tile = (t < 0);
    int kb = global_tile ? 0 : (qbase - WW + t * 64);
    if (!global_tile && (kb + 63 < 0 || kb >= SS)) continue;  // uniform per block

    // prefetch next band tile (lowers to global_prefetch_b8)
    {
      int pk = (global_tile ? qbase - WW : kb + 64) + (lane & 15) * 4;
      pk = pk < 0 ? 0 : (pk >= SS ? SS - 1 : pk);
      __builtin_prefetch(kh + (size_t)pk * DD, 0, 0);
      __builtin_prefetch(vh + (size_t)pk * DD, 0, 0);
    }

    __syncthreads();  // protect Vt reuse
    // stage V tile transposed: Vt[d][key]
#pragma unroll
    for (int c = 0; c < 4; ++c) {
      int lin = threadIdx.x * 4 + c;  // 0..511, 8 elems each
      int key = lin >> 3;
      int d0 = (lin & 7) * 8;
      int gk = kb + key;
      gk = gk < 0 ? 0 : (gk >= SS ? SS - 1 : gk);
      float4 tmp = *(const float4*)(vh + (size_t)gk * DD + d0);
      const unsigned short* ts = (const unsigned short*)&tmp;
#pragma unroll
      for (int i = 0; i < 8; ++i) Vt[d0 + i][key] = ts[i];
    }
    __syncthreads();

    // scores: 16 rows x 64 keys per wave
    v8f sc[4];
#pragma unroll
    for (int nt = 0; nt < 4; ++nt) {
      int jj = kb + nt * 16 + (lane & 15);
      int jcl = jj < 0 ? 0 : (jj >= SS ? SS - 1 : jj);
      const unsigned short* krow = kh + (size_t)jcl * DD;
      v16bf b0 = frag_load(krow, 0, lane);
      v16bf b1 = frag_load(krow, 32, lane);
      v8f s = z;
      s = WMMA_BF16(aq0, b0, s);
      s = WMMA_BF16(aq1, b1, s);
      if (!global_tile) {
#pragma unroll
        for (int r = 0; r < 8; ++r) {
          int i = qbase + wave * 16 + rbase + r;
          int dlt = jj - i;
          bool valid = (jj >= 0) && (jj < SS) && (dlt >= -WW) && (dlt <= WW);
          float add = (valid && jj < GG) ? -10000.f : 0.f;
          s[r] = valid ? (s[r] + add) : NEGV;
        }
      }
      sc[nt] = s;
    }

    // online softmax (row reductions: 4 accums then 16-lane shuffle group)
#pragma unroll
    for (int r = 0; r < 8; ++r) {
      float v0 = fmaxf(fmaxf(sc[0][r], sc[1][r]), fmaxf(sc[2][r], sc[3][r]));
#pragma unroll
      for (int off = 1; off < 16; off <<= 1) v0 = fmaxf(v0, __shfl_xor(v0, off, 32));
      float nm = fmaxf(m[r], v0);
      float alpha = __expf(m[r] - nm);
      m[r] = nm;
      l[r] *= alpha;
#pragma unroll
      for (int nt = 0; nt < 4; ++nt) o[nt][r] *= alpha;
    }
    int prow = wave * 16 + rbase;
    float psum[8];
#pragma unroll
    for (int r = 0; r < 8; ++r) psum[r] = 0.f;
#pragma unroll
    for (int nt = 0; nt < 4; ++nt) {
#pragma unroll
      for (int r = 0; r < 8; ++r) {
        float p = __expf(sc[nt][r] - m[r]);
        psum[r] += p;
        P[prow + r][nt * 16 + (lane & 15)] = f2bf(p);
      }
    }
#pragma unroll
    for (int r = 0; r < 8; ++r) {
      float s0 = psum[r];
#pragma unroll
      for (int off = 1; off < 16; off <<= 1) s0 += __shfl_xor(s0, off, 32);
      l[r] += s0;
    }

    // PV: o += P(16x64) @ V(64x64)  (same-wave LDS RAW is in-order)
    const unsigned short* prd = &P[wave * 16 + (lane & 15)][0];
    v16bf ap0 = frag_load(prd, 0, lane);
    v16bf ap1 = frag_load(prd, 32, lane);
#pragma unroll
    for (int nt = 0; nt < 4; ++nt) {
      const unsigned short* vrow = &Vt[nt * 16 + (lane & 15)][0];
      v16bf b0 = frag_load(vrow, 0, lane);
      v16bf b1 = frag_load(vrow, 32, lane);
      o[nt] = WMMA_BF16(ap0, b0, o[nt]);
      o[nt] = WMMA_BF16(ap1, b1, o[nt]);
    }
  }

#pragma unroll
  for (int nt = 0; nt < 4; ++nt) {
    int col = h * DD + nt * 16 + (lane & 15);
#pragma unroll
    for (int r = 0; r < 8; ++r) {
      int s = qbase + wave * 16 + rbase + r;
      out[(size_t)s * EE + col] = o[nt][r] / l[r];
    }
  }
}

// ---------------- global-row attention: rows 0..63, full 4096-key softmax ----------------
// grid: 12 blocks (one head each); block 128 (4 waves x 16 rows).
__global__ void __launch_bounds__(128) attn_globalrows(
    const unsigned short* __restrict__ Qg,  // [H][64][64] bf16 (pre-scaled)
    const unsigned short* __restrict__ Kg,  // [H][S][64]
    const unsigned short* __restrict__ Vg,  // [H][S][64]
    float* __restrict__ out) {
  __shared__ __align__(16) unsigned short Vt[64][64];
  __shared__ __align__(16) unsigned short P[64][64];

  int lane = threadIdx.x & 31, wave = threadIdx.x >> 5;
  int h = blockIdx.x;
  const unsigned short* qh = Qg + (size_t)h * GG * DD;
  const unsigned short* kh = Kg + (size_t)h * SS * DD;
  const unsigned short* vh = Vg + (size_t)h * SS * DD;

  const unsigned short* qrow = qh + (size_t)(wave * 16 + (lane & 15)) * DD;
  v16bf aq0 = frag_load(qrow, 0, lane);
  v16bf aq1 = frag_load(qrow, 32, lane);

  v8f z = {0.f, 0.f, 0.f, 0.f, 0.f, 0.f, 0.f, 0.f};
  v8f o[4] = {z, z, z, z};
  float m[8], l[8];
#pragma unroll
  for (int r = 0; r < 8; ++r) { m[r] = -1e30f; l[r] = 0.f; }
  int rbase = (lane >> 4) * 8;

  for (int t = 0; t < SS / 64; ++t) {
    int kb = t * 64;
    // prefetch next tile
    {
      int pk = kb + 64 + (lane & 15) * 4;
      pk = pk >= SS ? SS - 1 : pk;
      __builtin_prefetch(kh + (size_t)pk * DD, 0, 0);
      __builtin_prefetch(vh + (size_t)pk * DD, 0, 0);
    }
    __syncthreads();
#pragma unroll
    for (int c = 0; c < 4; ++c) {
      int lin = threadIdx.x * 4 + c;
      int key = lin >> 3;
      int d0 = (lin & 7) * 8;
      float4 tmp = *(const float4*)(vh + (size_t)(kb + key) * DD + d0);
      const unsigned short* ts = (const unsigned short*)&tmp;
#pragma unroll
      for (int i = 0; i < 8; ++i) Vt[d0 + i][key] = ts[i];
    }
    __syncthreads();

    v8f sc[4];
#pragma unroll
    for (int nt = 0; nt < 4; ++nt) {
      int jj = kb + nt * 16 + (lane & 15);
      const unsigned short* krow = kh + (size_t)jj * DD;
      v16bf b0 = frag_load(krow, 0, lane);
      v16bf b1 = frag_load(krow, 32, lane);
      v8f s = z;
      s = WMMA_BF16(aq0, b0, s);
      s = WMMA_BF16(aq1, b1, s);
      sc[nt] = s;
    }

#pragma unroll
    for (int r = 0; r < 8; ++r) {
      float v0 = fmaxf(fmaxf(sc[0][r], sc[1][r]), fmaxf(sc[2][r], sc[3][r]));
#pragma unroll
      for (int off = 1; off < 16; off <<= 1) v0 = fmaxf(v0, __shfl_xor(v0, off, 32));
      float nm = fmaxf(m[r], v0);
      float alpha = __expf(m[r] - nm);
      m[r] = nm;
      l[r] *= alpha;
#pragma unroll
      for (int nt = 0; nt < 4; ++nt) o[nt][r] *= alpha;
    }
    int prow = wave * 16 + rbase;
    float psum[8];
#pragma unroll
    for (int r = 0; r < 8; ++r) psum[r] = 0.f;
#pragma unroll
    for (int nt = 0; nt < 4; ++nt) {
#pragma unroll
      for (int r = 0; r < 8; ++r) {
        float p = __expf(sc[nt][r] - m[r]);
        psum[r] += p;
        P[prow + r][nt * 16 + (lane & 15)] = f2bf(p);
      }
    }
#pragma unroll
    for (int r = 0; r < 8; ++r) {
      float s0 = psum[r];
#pragma unroll
      for (int off = 1; off < 16; off <<= 1) s0 += __shfl_xor(s0, off, 32);
      l[r] += s0;
    }

    const unsigned short* prd = &P[wave * 16 + (lane & 15)][0];
    v16bf ap0 = frag_load(prd, 0, lane);
    v16bf ap1 = frag_load(prd, 32, lane);
#pragma unroll
    for (int nt = 0; nt < 4; ++nt) {
      const unsigned short* vrow = &Vt[nt * 16 + (lane & 15)][0];
      v16bf b0 = frag_load(vrow, 0, lane);
      v16bf b1 = frag_load(vrow, 32, lane);
      o[nt] = WMMA_BF16(ap0, b0, o[nt]);
      o[nt] = WMMA_BF16(ap1, b1, o[nt]);
    }
  }

#pragma unroll
  for (int nt = 0; nt < 4; ++nt) {
    int col = h * DD + nt * 16 + (lane & 15);
#pragma unroll
    for (int r = 0; r < 8; ++r) {
      int s = wave * 16 + rbase + r;  // rows 0..63 overwrite band output
      out[(size_t)s * EE + col] = o[nt][r] / l[r];
    }
  }
}

// ---------------- launch ----------------
extern "C" void kernel_launch(void* const* d_in, const int* in_sizes, int n_in,
                              void* d_out, int out_size, void* d_ws, size_t ws_size,
                              hipStream_t stream) {
  (void)in_sizes; (void)n_in; (void)out_size; (void)ws_size;

  const float* hidden = (const float*)d_in[0];
  const float* Wq  = (const float*)d_in[1];  const float* bq  = (const float*)d_in[2];
  const float* Wk  = (const float*)d_in[3];  const float* bk  = (const float*)d_in[4];
  const float* Wv  = (const float*)d_in[5];  const float* bv  = (const float*)d_in[6];
  const float* Wqg = (const float*)d_in[7];  const float* bqg = (const float*)d_in[8];
  const float* Wkg = (const float*)d_in[9];  const float* bkg = (const float*)d_in[10];
  const float* Wvg = (const float*)d_in[11]; const float* bvg = (const float*)d_in[12];
  float* out = (float*)d_out;

  // Workspace layout (bf16 ushorts); ~45 MB total.
  unsigned short* ws = (unsigned short*)d_ws;
  size_t off = 0;
  auto alloc = [&](size_t n) { unsigned short* p = ws + off; off += n; return p; };
  unsigned short* hb   = alloc((size_t)SS * EE);
  unsigned short* Wtq  = alloc((size_t)EE * EE);
  unsigned short* Wtk  = alloc((size_t)EE * EE);
  unsigned short* Wtv  = alloc((size_t)EE * EE);
  unsigned short* Wtqg = alloc((size_t)EE * EE);
  unsigned short* Wtkg = alloc((size_t)EE * EE);
  unsigned short* Wtvg = alloc((size_t)EE * EE);
  unsigned short* qB   = alloc((size_t)HH * SS * DD);
  unsigned short* kB   = alloc((size_t)HH * SS * DD);
  unsigned short* vB   = alloc((size_t)HH * SS * DD);
  unsigned short* kgB  = alloc((size_t)HH * SS * DD);
  unsigned short* vgB  = alloc((size_t)HH * SS * DD);
  unsigned short* qgB  = alloc((size_t)HH * GG * DD);

  const float scale = 0.125f;  // 1/sqrt(64)

  cvt_f32_bf16<<<(SS * EE + 255) / 256, 256, 0, stream>>>(hidden, hb, SS * EE);
  int wgrid = (EE * EE + 255) / 256;
  cvt_weight_t<<<wgrid, 256, 0, stream>>>(Wq,  Wtq);
  cvt_weight_t<<<wgrid, 256, 0, stream>>>(Wk,  Wtk);
  cvt_weight_t<<<wgrid, 256, 0, stream>>>(Wv,  Wtv);
  cvt_weight_t<<<wgrid, 256, 0, stream>>>(Wqg, Wtqg);
  cvt_weight_t<<<wgrid, 256, 0, stream>>>(Wkg, Wtkg);
  cvt_weight_t<<<wgrid, 256, 0, stream>>>(Wvg, Wtvg);

  gemm_proj<<<dim3(SS / 128, HH), 128, 0, stream>>>(hb, Wtq,  bq,  qB,  SS, scale);
  gemm_proj<<<dim3(SS / 128, HH), 128, 0, stream>>>(hb, Wtk,  bk,  kB,  SS, 1.0f);
  gemm_proj<<<dim3(SS / 128, HH), 128, 0, stream>>>(hb, Wtv,  bv,  vB,  SS, 1.0f);
  gemm_proj<<<dim3(SS / 128, HH), 128, 0, stream>>>(hb, Wtkg, bkg, kgB, SS, 1.0f);
  gemm_proj<<<dim3(SS / 128, HH), 128, 0, stream>>>(hb, Wtvg, bvg, vgB, SS, 1.0f);
  gemm_proj<<<dim3(1,       HH), 128, 0, stream>>>(hb, Wtqg, bqg, qgB, GG, scale);

  attn_band<<<dim3(SS / 64, HH), 128, 0, stream>>>(qB, kB, vB, out);
  attn_globalrows<<<HH, 128, 0, stream>>>(qgB, kgB, vgB, out);  // overwrites rows 0..g-1
}
